// LinearEncoder_53480932769970
// MI455X (gfx1250) — matrix operand order repaired and checked by reading, not compiled, not run on hardware
//
#include <hip/hip_runtime.h>
#include <hip/hip_bf16.h>
#include <stdint.h>

typedef __attribute__((ext_vector_type(16))) __bf16 v16bf;
typedef __attribute__((ext_vector_type(8)))  float  v8f;
typedef __attribute__((ext_vector_type(4)))  int    v4i;

#define EPSF 1e-5f
#define MB 8192   /* B*L rows */
#define KD 512    /* D        */
#define NE 8192   /* D*T cols */

#ifndef USE_ASYNC
#if __has_builtin(__builtin_amdgcn_global_load_async_to_lds_b128)
#define USE_ASYNC 1
#else
#define USE_ASYNC 0
#endif
#endif

#if USE_ASYNC
#define ASYNC_CP16(gp, lp) \
  __builtin_amdgcn_global_load_async_to_lds_b128( \
      (__attribute__((address_space(1))) v4i*)(gp), \
      (__attribute__((address_space(3))) v4i*)(lp), 0, 0)
#endif

// ---------------- LayerNorm over D + bf16 pack of A ----------------
__global__ __launch_bounds__(256) void ln_pack_kernel(
    const float* __restrict__ x, const float* __restrict__ g,
    const float* __restrict__ bta, __hip_bfloat16* __restrict__ outA)
{
  __shared__ float ssum[256], ssq[256];
  const int row = blockIdx.x;
  const int tid = threadIdx.x;
  const float2 xv = *(const float2*)(x + (size_t)row * KD + 2 * tid);
  ssum[tid] = xv.x + xv.y;
  ssq[tid]  = xv.x * xv.x + xv.y * xv.y;
  __syncthreads();
  for (int s = 128; s > 0; s >>= 1) {
    if (tid < s) { ssum[tid] += ssum[tid + s]; ssq[tid] += ssq[tid + s]; }
    __syncthreads();
  }
  const float mean = ssum[0] * (1.0f / KD);
  const float var  = ssq[0] * (1.0f / KD) - mean * mean;
  const float rstd = rsqrtf(var + EPSF);
  const int d0 = 2 * tid;
  const float y0 = (xv.x - mean) * rstd * g[d0]     + bta[d0];
  const float y1 = (xv.y - mean) * rstd * g[d0 + 1] + bta[d0 + 1];
  union { __hip_bfloat16 h[2]; unsigned int u; } p;
  p.h[0] = __float2bfloat16(y0);
  p.h[1] = __float2bfloat16(y1);
  *(unsigned int*)((unsigned short*)outA + (size_t)row * KD + d0) = p.u;
}

// ---------------- W f32 -> bf16 pack (already (N,K) row-major) ----------------
__global__ __launch_bounds__(256) void wpack_kernel(
    const float* __restrict__ W, __hip_bfloat16* __restrict__ Wb)
{
  const size_t i = (size_t)blockIdx.x * 256 + threadIdx.x;
  const float4 w = *(const float4*)(W + 4 * i);
  union { __hip_bfloat16 h[4]; uint2 u; } p;
  p.h[0] = __float2bfloat16(w.x); p.h[1] = __float2bfloat16(w.y);
  p.h[2] = __float2bfloat16(w.z); p.h[3] = __float2bfloat16(w.w);
  *(uint2*)((unsigned short*)Wb + 4 * i) = p.u;
}

__global__ void zero_stats_kernel(float* stats)
{
  if (threadIdx.x < 32) stats[threadIdx.x] = 0.0f;
}

// ---------------- WMMA GEMM + bias + BN-stat epilogue ----------------
#define LDSW 40                 /* ushorts per tile row: 32 data + 8 pad (80B, 16B aligned) */
#define TILE_USH (128 * LDSW)   /* 5120 ushorts per 128x32 bf16 tile */

__global__ __launch_bounds__(256) void gemm_kernel(
    const __hip_bfloat16* __restrict__ A, const __hip_bfloat16* __restrict__ Bw,
    const float* __restrict__ bias, float* __restrict__ C, float* __restrict__ stats)
{
  __shared__ __align__(16) unsigned short smem[2 * 2 * TILE_USH];  // [buf][A|B][tile]
  __shared__ float sred[32];

  const int tid  = threadIdx.x;
  const int lane = tid & 31;
  const int wave = tid >> 5;
  const int wm = wave >> 2, wn = wave & 3;   // 2x4 wave grid -> 64x32 per wave
  const int bm = blockIdx.y, bn = blockIdx.x;

  // global->LDS chunk assignment: 512 16B-chunks per tile, 2 per thread per tile
  const int r0 = tid >> 2,          c0 = tid & 3;
  const int r1 = (tid + 256) >> 2,  c1 = (tid + 256) & 3;

  const unsigned short* gA = (const unsigned short*)A + (size_t)(bm * 128) * KD;
  const unsigned short* gB = (const unsigned short*)Bw + (size_t)(bn * 128) * KD;

  v8f acc[4][2];
#pragma unroll
  for (int i = 0; i < 4; i++)
#pragma unroll
    for (int j = 0; j < 2; j++) acc[i][j] = (v8f)(0.0f);

  auto issue = [&](int ks, int buf) {
    const int k0 = ks * 32;
    unsigned short* la = smem + buf * (2 * TILE_USH);
    unsigned short* lb = la + TILE_USH;
#if USE_ASYNC
    ASYNC_CP16(gA + (size_t)r0 * KD + k0 + c0 * 8, la + r0 * LDSW + c0 * 8);
    ASYNC_CP16(gA + (size_t)r1 * KD + k0 + c1 * 8, la + r1 * LDSW + c1 * 8);
    ASYNC_CP16(gB + (size_t)r0 * KD + k0 + c0 * 8, lb + r0 * LDSW + c0 * 8);
    ASYNC_CP16(gB + (size_t)r1 * KD + k0 + c1 * 8, lb + r1 * LDSW + c1 * 8);
#else
    *(uint4*)(la + r0 * LDSW + c0 * 8) = *(const uint4*)(gA + (size_t)r0 * KD + k0 + c0 * 8);
    *(uint4*)(la + r1 * LDSW + c1 * 8) = *(const uint4*)(gA + (size_t)r1 * KD + k0 + c1 * 8);
    *(uint4*)(lb + r0 * LDSW + c0 * 8) = *(const uint4*)(gB + (size_t)r0 * KD + k0 + c0 * 8);
    *(uint4*)(lb + r1 * LDSW + c1 * 8) = *(const uint4*)(gB + (size_t)r1 * KD + k0 + c1 * 8);
#endif
  };

  const int halem = lane >> 4;   // K-half selector
  const int nlo   = lane & 15;   // row (A) / col (B,C)

  auto compute = [&](int buf) {
    const unsigned short* la = smem + buf * (2 * TILE_USH);
    const unsigned short* lb = la + TILE_USH;
    union FU { v16bf v; uint4 q[2]; };
    FU af[4], bfr[2];
#pragma unroll
    for (int i = 0; i < 4; i++) {
      const unsigned short* p = la + (wm * 64 + i * 16 + nlo) * LDSW + halem * 8;
      af[i].q[0] = *(const uint4*)p;          // K = h*8 .. h*8+7
      af[i].q[1] = *(const uint4*)(p + 16);   // K = h*8+16 .. h*8+23
    }
#pragma unroll
    for (int j = 0; j < 2; j++) {
      const unsigned short* p = lb + (wn * 32 + j * 16 + nlo) * LDSW + halem * 8;
      bfr[j].q[0] = *(const uint4*)p;         // K = h*8 .. h*8+7
      bfr[j].q[1] = *(const uint4*)(p + 16);  // K = h*8+16 .. h*8+23
    }
#pragma unroll
    for (int i = 0; i < 4; i++)
#pragma unroll
      for (int j = 0; j < 2; j++)
        acc[i][j] = __builtin_amdgcn_wmma_f32_16x16x32_bf16(
            false, af[i].v, false, bfr[j].v, (short)0, acc[i][j], false, false);
  };

#if USE_ASYNC
  issue(0, 0);
  for (int ks = 0; ks < 16; ++ks) {
    asm volatile("s_wait_asynccnt 0x0" ::: "memory");  // own tile-ks loads complete
    __syncthreads();                                   // all waves loaded ks; all done computing ks-1
    if (ks + 1 < 16) issue(ks + 1, (ks + 1) & 1);      // DMA next tile overlaps compute
    compute(ks & 1);
  }
#else
  for (int ks = 0; ks < 16; ++ks) {
    __syncthreads();
    issue(ks, 0);
    __syncthreads();
    compute(0);
  }
#endif

  // ---- epilogue: bias, store C, per-channel (t = n%16 = nlo) sum/sumsq ----
  if (tid < 32) sred[tid] = 0.0f;
  __syncthreads();

  float bsum = 0.0f, bsq = 0.0f;
#pragma unroll
  for (int j = 0; j < 2; j++) {
    const int n = bn * 128 + wn * 32 + j * 16 + nlo;
    const float bb = bias[n];
#pragma unroll
    for (int i = 0; i < 4; i++) {
      const int m0 = bm * 128 + wm * 64 + i * 16 + halem * 8;
      float* crow = C + (size_t)m0 * NE + n;
      v8f a = acc[i][j];
#pragma unroll
      for (int r = 0; r < 8; r++) {
        const float val = a[r] + bb;
        bsum += val;
        bsq  += val * val;
        crow[(size_t)r * NE] = val;
      }
    }
  }
  atomicAdd(&sred[nlo], bsum);
  atomicAdd(&sred[16 + nlo], bsq);
  __syncthreads();
  if (tid < 32) atomicAdd(&stats[tid], sred[tid]);
}

// ---------------- BN coefficient finalize ----------------
__global__ void bnfin_kernel(const float* __restrict__ stats,
                             const float* __restrict__ bn_g,
                             const float* __restrict__ bn_b,
                             float* __restrict__ coef)
{
  const int t = threadIdx.x;
  if (t < 16) {
    const float cnt = (float)MB * (float)(NE / 16);   // 8192 * 512 per channel
    const float mean = stats[t] / cnt;
    const float var  = stats[16 + t] / cnt - mean * mean;
    const float sc   = bn_g[t] * rsqrtf(var + EPSF);
    coef[t]      = sc;
    coef[16 + t] = bn_b[t] - mean * sc;
  }
}

// ---------------- fused BN + multi-step LIF ----------------
__global__ __launch_bounds__(256) void lif_kernel(
    const float* __restrict__ C, const float* __restrict__ coef,
    const float* __restrict__ th, float* __restrict__ out)
{
  __shared__ float sc[32];
  __shared__ float sth;
  if (threadIdx.x < 32) sc[threadIdx.x] = coef[threadIdx.x];
  if (threadIdx.x == 0) sth = th[0];
  __syncthreads();

  const size_t u = (size_t)blockIdx.x * 256 + threadIdx.x;  // u = m*512 + d
  const float4* src = (const float4*)(C + u * 16);          // C[m][d*16 + t], t contiguous
  float xs[16];
#pragma unroll
  for (int q = 0; q < 4; q++) {
    const float4 v4 = src[q];
    xs[4 * q + 0] = v4.x; xs[4 * q + 1] = v4.y;
    xs[4 * q + 2] = v4.z; xs[4 * q + 3] = v4.w;
  }
  const float thr = sth;
  float v = 0.0f;
  const size_t plane = (size_t)MB * KD;                      // 4,194,304
#pragma unroll
  for (int t = 0; t < 16; t++) {
    const float x = xs[t] * sc[t] + sc[16 + t];
    v = 0.5f * (v + x);                                      // v += (x - v)/tau, tau=2
    const float s = (v >= thr) ? 1.0f : 0.0f;                // heaviside
    __builtin_nontemporal_store(s, out + (size_t)t * plane + u);
    v = (1.0f - s) * v;                                      // hard reset, detached
  }
}

// ---------------- host ----------------
extern "C" void kernel_launch(void* const* d_in, const int* in_sizes, int n_in,
                              void* d_out, int out_size, void* d_ws, size_t ws_size,
                              hipStream_t stream)
{
  (void)in_sizes; (void)n_in; (void)out_size; (void)ws_size;
  const float* inputs = (const float*)d_in[0];
  const float* W      = (const float*)d_in[1];
  const float* bias   = (const float*)d_in[2];
  const float* ln_g   = (const float*)d_in[3];
  const float* ln_b   = (const float*)d_in[4];
  const float* bn_g   = (const float*)d_in[5];
  const float* bn_b   = (const float*)d_in[6];
  const float* th     = (const float*)d_in[7];
  float* out = (float*)d_out;

  char* ws = (char*)d_ws;
  __hip_bfloat16* Abf = (__hip_bfloat16*)(ws);                                  //  8 MB
  __hip_bfloat16* Wbf = (__hip_bfloat16*)(ws + (size_t)MB * KD * 2);            //  8 MB
  float* C     = (float*)(ws + (size_t)MB * KD * 2 + (size_t)NE * KD * 2);      // 256 MB
  float* stats = (float*)((char*)C + (size_t)MB * NE * 4);                      // 32 f
  float* coef  = stats + 32;                                                    // 32 f

  zero_stats_kernel<<<1, 32, 0, stream>>>(stats);
  ln_pack_kernel<<<MB, 256, 0, stream>>>(inputs, ln_g, ln_b, Abf);
  wpack_kernel<<<(NE * KD / 4) / 256, 256, 0, stream>>>(W, Wbf);
  gemm_kernel<<<dim3(NE / 128, MB / 128), 256, 0, stream>>>(Abf, Wbf, bias, C, stats);
  bnfin_kernel<<<1, 32, 0, stream>>>(stats, bn_g, bn_b, coef);
  lif_kernel<<<(MB * KD) / 256, 256, 0, stream>>>(C, coef, th, out);
}
